// MTCNN_89610197664184
// MI455X (gfx1250) — matrix operand order repaired and checked by reading, not compiled, not run on hardware
//
#include <hip/hip_runtime.h>
#include <hip/hip_bf16.h>

typedef __attribute__((ext_vector_type(16))) _Float16 v16h;
typedef __attribute__((ext_vector_type(8)))  _Float16 v8h;
typedef __attribute__((ext_vector_type(8)))  float    v8f;

#define NB   16          // batch
#define W0   720         // input H/W
#define P1   359         // after conv1(718) + maxpool2 -> 359
#define W2O  357         // conv2 output H/W
#define W3O  355         // conv3 output H/W
#define HW3  (355*355)

// ---------------------------------------------------------------------------
// Kernel A: fused normalize + conv1(3->10,3x3,VALID) + PReLU + maxpool2x2.
// K=27 is too thin for WMMA -> VALU FMA. One thread per pooled pixel;
// weight indices are loop-uniform -> scalar loads. Output f16 NCHW feeds
// conv2's WMMA A operands without conversion.
// ---------------------------------------------------------------------------
__global__ void k_conv1(const float* __restrict__ x, const float* __restrict__ w,
                        const float* __restrict__ bias, const float* __restrict__ alpha,
                        _Float16* __restrict__ out) {
  int idx = blockIdx.x * blockDim.x + threadIdx.x;
  const int NP = NB * P1 * P1;
  if (idx >= NP) return;
  int px = idx % P1;
  int py = (idx / P1) % P1;
  int b  = idx / (P1 * P1);

  float v[3][4][4];
  #pragma unroll
  for (int c = 0; c < 3; ++c)
    #pragma unroll
    for (int dy = 0; dy < 4; ++dy)
      #pragma unroll
      for (int dx = 0; dx < 4; ++dx) {
        int iy = 2 * py + dy, ix = 2 * px + dx;     // max 719 < 720
        float t = x[((size_t)(b * 3 + c) * W0 + iy) * W0 + ix];
        v[c][dy][dx] = (t - 127.5f) * 0.0078125f;
      }

  for (int co = 0; co < 10; ++co) {
    float bco = bias[co], aco = alpha[co];
    float m = -3.0e38f;
    #pragma unroll
    for (int sy = 0; sy < 2; ++sy)
      #pragma unroll
      for (int sx = 0; sx < 2; ++sx) {
        float acc = bco;
        #pragma unroll
        for (int c = 0; c < 3; ++c)
          #pragma unroll
          for (int kh = 0; kh < 3; ++kh)
            #pragma unroll
            for (int kw = 0; kw < 3; ++kw)
              acc = fmaf(w[((co * 3 + c) * 3 + kh) * 3 + kw], v[c][sy + kh][sx + kw], acc);
        acc = acc >= 0.f ? acc : aco * acc;          // PReLU
        m = fmaxf(m, acc);                           // maxpool 2x2
      }
    out[((size_t)(b * 10 + co) * P1 + py) * P1 + px] = (_Float16)m;
  }
}

// ---------------------------------------------------------------------------
// Kernel B: conv2 (10->16,3x3) implicit GEMM, v_wmma_f32_16x16x32_f16.
// One wave per output row: builds the 3 weight (B) fragments once, then
// loops over 23 tiles of 16 pixels. The input tile is staged in LDS in
// im2col layout col[M][96] (K = ci*9+kh*3+kw, ci-major, pad 90->96) so each
// A fragment is two aligned ds_load_b128:
//   bytes = M*192 + kb*64 + g*16  and +32   (matches the 16-bit A layout:
//   V0-3 hold K 0-7 (g0) / 8-15 (g1), V4-7 hold K 16-23 / 24-31).
// All K indices are compile-time constants -> no division, no branches.
// ---------------------------------------------------------------------------
__global__ void k_conv2(const _Float16* __restrict__ in, const float* __restrict__ w,
                        const float* __restrict__ bias, const float* __restrict__ alpha,
                        _Float16* __restrict__ out) {
  const int l  = threadIdx.x;            // 32 threads = 1 wave
  const int g  = (l >> 4) & 1;
  const int n  = l & 15;                 // A-row (pixel) / B-col (cout) for this lane
  const int oy = blockIdx.x;
  const int b  = blockIdx.y;

  __shared__ _Float16 col[16 * 96];      // im2col tile, 3 KB

  // --- B (weight) fragments, built once: lane = cout n, K = kb*32+16(v>>2)+8g+2(v&3)+p ---
  v16h Bf[3];
  #pragma unroll
  for (int kb = 0; kb < 3; ++kb)
    #pragma unroll
    for (int v = 0; v < 8; ++v)
      #pragma unroll
      for (int p = 0; p < 2; ++p) {
        int K = kb * 32 + (v >> 2) * 16 + g * 8 + (v & 3) * 2 + p;
        bool valid = K < 90;                       // folds for kb<2 (g in {0,1})
        float wv = w[valid ? (n * 90 + K) : 0];    // clamped load, no branch
        Bf[kb][v * 2 + p] = valid ? (_Float16)wv : (_Float16)0.f;
      }

  const size_t inb = (size_t)b * 10 * P1 * P1;
  const float bn = bias[n], an = alpha[n];

  for (int t = 0; t < 23; ++t) {
    const int ox0 = t * 16;

    // --- im2col fill: lane n fills half (by g) of its own K-column ---
    #pragma unroll
    for (int cc = 0; cc < 5; ++cc) {
      const int ci = g * 5 + cc;
      #pragma unroll
      for (int kh = 0; kh < 3; ++kh) {
        const _Float16* rowp = in + inb + ((size_t)ci * P1 + (oy + kh)) * P1;
        #pragma unroll
        for (int kw = 0; kw < 3; ++kw) {
          int ix = ox0 + n + kw;
          ix = ix < (P1 - 1) ? ix : (P1 - 1);      // clamp (tail tile), no branch
          col[n * 96 + ci * 9 + kh * 3 + kw] = rowp[ix];
        }
      }
    }
    // zero pad K = 90..95 (bytes n*192+180..191, dword aligned)
    {
      uint32_t* zp = (uint32_t*)&col[n * 96 + 90];
      zp[0] = 0u; zp[1] = 0u; zp[2] = 0u;
    }
    __syncthreads();

    // --- 3 K-steps of WMMA; A fragments via 2x ds_load_b128 each ---
    v8f acc = {};
    #pragma unroll
    for (int kb = 0; kb < 3; ++kb) {
      const v8h* pl = (const v8h*)&col[n * 96 + g * 8 + kb * 32];
      v8h alo = pl[0];                   // +0  bytes: K 0-7 (g0) / 8-15 (g1)
      v8h ahi = pl[2];                   // +32 bytes: K 16-23 / 24-31
      v16h Af = __builtin_shufflevector(alo, ahi,
                 0,1,2,3,4,5,6,7,8,9,10,11,12,13,14,15);
      acc = __builtin_amdgcn_wmma_f32_16x16x32_f16(false, Af, false, Bf[kb],
                                                   (short)0, acc, false, false);
    }
    __syncthreads();

    // --- bias + PReLU + store: lane -> cout n, VGPR r -> pixel r+8g ---
    #pragma unroll
    for (int r = 0; r < 8; ++r) {
      int ox = ox0 + r + 8 * g;
      if (ox < W2O) {
        float y = acc[r] + bn;
        y = y >= 0.f ? y : an * y;
        out[((size_t)(b * 16 + n) * W2O + oy) * W2O + ox] = (_Float16)y;
      }
    }
  }
}

// ---------------------------------------------------------------------------
// Kernel C: conv3 (16->32,3x3). K=144 pad->160 (5 K-steps), N=32 -> two
// N-tiles -> 10 WMMAs per 16-pixel tile. Same im2col scheme, col[M][160].
// ---------------------------------------------------------------------------
__global__ void k_conv3(const _Float16* __restrict__ in, const float* __restrict__ w,
                        const float* __restrict__ bias, const float* __restrict__ alpha,
                        _Float16* __restrict__ out) {
  const int l  = threadIdx.x;
  const int g  = (l >> 4) & 1;
  const int n  = l & 15;
  const int oy = blockIdx.x;
  const int b  = blockIdx.y;

  __shared__ _Float16 col[16 * 160];     // 5 KB

  v16h Bf[5][2];
  #pragma unroll
  for (int kb = 0; kb < 5; ++kb)
    #pragma unroll
    for (int nt = 0; nt < 2; ++nt)
      #pragma unroll
      for (int v = 0; v < 8; ++v)
        #pragma unroll
        for (int p = 0; p < 2; ++p) {
          int K = kb * 32 + (v >> 2) * 16 + g * 8 + (v & 3) * 2 + p;
          bool valid = K < 144;                       // folds for kb<4
          float wv = w[valid ? ((n + 16 * nt) * 144 + K) : 0];
          Bf[kb][nt][v * 2 + p] = valid ? (_Float16)wv : (_Float16)0.f;
        }

  const size_t inb = (size_t)b * 16 * W2O * W2O;
  const float b0 = bias[n],      a0 = alpha[n];
  const float b1 = bias[n + 16], a1 = alpha[n + 16];

  for (int t = 0; t < 23; ++t) {
    const int ox0 = t * 16;

    #pragma unroll
    for (int cc = 0; cc < 8; ++cc) {
      const int ci = g * 8 + cc;
      #pragma unroll
      for (int kh = 0; kh < 3; ++kh) {
        const _Float16* rowp = in + inb + ((size_t)ci * W2O + (oy + kh)) * W2O;
        #pragma unroll
        for (int kw = 0; kw < 3; ++kw) {
          int ix = ox0 + n + kw;
          ix = ix < (W2O - 1) ? ix : (W2O - 1);
          col[n * 160 + ci * 9 + kh * 3 + kw] = rowp[ix];
        }
      }
    }
    // zero pad K = 144..159: two aligned b128 stores (bytes n*320+288, +304)
    {
      v8h z = {};
      v8h* zp = (v8h*)&col[n * 160 + 144];
      zp[0] = z; zp[1] = z;
    }
    __syncthreads();

    v8f acc0 = {}, acc1 = {};
    #pragma unroll
    for (int kb = 0; kb < 5; ++kb) {
      const v8h* pl = (const v8h*)&col[n * 160 + g * 8 + kb * 32];
      v8h alo = pl[0];
      v8h ahi = pl[2];
      v16h Af = __builtin_shufflevector(alo, ahi,
                 0,1,2,3,4,5,6,7,8,9,10,11,12,13,14,15);
      acc0 = __builtin_amdgcn_wmma_f32_16x16x32_f16(false, Af, false, Bf[kb][0],
                                                    (short)0, acc0, false, false);
      acc1 = __builtin_amdgcn_wmma_f32_16x16x32_f16(false, Af, false, Bf[kb][1],
                                                    (short)0, acc1, false, false);
    }
    __syncthreads();

    #pragma unroll
    for (int r = 0; r < 8; ++r) {
      int ox = ox0 + r + 8 * g;
      if (ox < W3O) {
        float y0 = acc0[r] + b0;  y0 = y0 >= 0.f ? y0 : a0 * y0;
        float y1 = acc1[r] + b1;  y1 = y1 >= 0.f ? y1 : a1 * y1;
        out[((size_t)(b * 32 + n)      * W3O + oy) * W3O + ox] = (_Float16)y0;
        out[((size_t)(b * 32 + n + 16) * W3O + oy) * W3O + ox] = (_Float16)y1;
      }
    }
  }
}

// ---------------------------------------------------------------------------
// Kernel D: fused 1x1 heads. prob = softmax(conv41), reg = conv42.
// Streaming (~130 MB read, 48 MB write); weights loop-uniform -> scalar.
// d_out = [reg 16*4*355*355] ++ [prob 16*2*355*355] (reference order).
// ---------------------------------------------------------------------------
__global__ void k_head(const _Float16* __restrict__ h3,
                       const float* __restrict__ w41, const float* __restrict__ b41,
                       const float* __restrict__ w42, const float* __restrict__ b42,
                       float* __restrict__ out) {
  int idx = blockIdx.x * blockDim.x + threadIdx.x;
  const int NT = NB * HW3;
  if (idx >= NT) return;
  int p = idx % HW3;
  int b = idx / HW3;

  float s0 = b41[0], s1 = b41[1];
  float r0 = b42[0], r1 = b42[1], r2 = b42[2], r3 = b42[3];
  #pragma unroll
  for (int c = 0; c < 32; ++c) {
    float h = (float)h3[(size_t)(b * 32 + c) * HW3 + p];
    s0 = fmaf(w41[c],      h, s0);
    s1 = fmaf(w41[32 + c], h, s1);
    r0 = fmaf(w42[c],      h, r0);
    r1 = fmaf(w42[32 + c], h, r1);
    r2 = fmaf(w42[64 + c], h, r2);
    r3 = fmaf(w42[96 + c], h, r3);
  }
  float mx = fmaxf(s0, s1);
  float e0 = __expf(s0 - mx), e1 = __expf(s1 - mx);
  float inv = 1.0f / (e0 + e1);

  const size_t REG = (size_t)NB * 4 * HW3;
  out[((size_t)(b * 4 + 0)) * HW3 + p] = r0;
  out[((size_t)(b * 4 + 1)) * HW3 + p] = r1;
  out[((size_t)(b * 4 + 2)) * HW3 + p] = r2;
  out[((size_t)(b * 4 + 3)) * HW3 + p] = r3;
  out[REG + ((size_t)(b * 2 + 0)) * HW3 + p] = e0 * inv;
  out[REG + ((size_t)(b * 2 + 1)) * HW3 + p] = e1 * inv;
}

static inline size_t alignup256(size_t v) { return (v + 255) & ~(size_t)255; }

extern "C" void kernel_launch(void* const* d_in, const int* in_sizes, int n_in,
                              void* d_out, int out_size, void* d_ws, size_t ws_size,
                              hipStream_t stream) {
  (void)in_sizes; (void)n_in; (void)out_size; (void)ws_size;
  const float* x    = (const float*)d_in[0];
  const float* w1   = (const float*)d_in[1];
  const float* b1   = (const float*)d_in[2];
  const float* a1   = (const float*)d_in[3];
  const float* w2   = (const float*)d_in[4];
  const float* b2   = (const float*)d_in[5];
  const float* a2   = (const float*)d_in[6];
  const float* w3   = (const float*)d_in[7];
  const float* b3   = (const float*)d_in[8];
  const float* a3   = (const float*)d_in[9];
  const float* w41  = (const float*)d_in[10];
  const float* b41  = (const float*)d_in[11];
  const float* w42  = (const float*)d_in[12];
  const float* b42  = (const float*)d_in[13];
  float* out = (float*)d_out;

  // workspace partition (all f16 NCHW): h1pooled | h2 | h3  (~236 MB)
  char* ws = (char*)d_ws;
  size_t off = 0;
  _Float16* h1p = (_Float16*)(ws + off);
  off += alignup256((size_t)NB * 10 * P1  * P1  * sizeof(_Float16));
  _Float16* h2  = (_Float16*)(ws + off);
  off += alignup256((size_t)NB * 16 * W2O * W2O * sizeof(_Float16));
  _Float16* h3  = (_Float16*)(ws + off);

  {
    int np = NB * P1 * P1;
    k_conv1<<<(np + 255) / 256, 256, 0, stream>>>(x, w1, b1, a1, h1p);
  }
  {
    dim3 grid(W2O, NB);                 // one wave per output row, 23 tiles each
    k_conv2<<<grid, 32, 0, stream>>>(h1p, w2, b2, a2, h2);
  }
  {
    dim3 grid(W3O, NB);
    k_conv3<<<grid, 32, 0, stream>>>(h2, w3, b3, a3, h3);
  }
  {
    int nt = NB * HW3;
    k_head<<<(nt + 255) / 256, 256, 0, stream>>>(h3, w41, b41, w42, b42, out);
  }
}
